// ConvolutionDMax_34076270526490
// MI455X (gfx1250) — matrix operand order repaired and checked by reading, not compiled, not run on hardware
//
#include <hip/hip_runtime.h>
#include <hip/hip_bf16.h>
#include <math.h>

// ---------------------------------------------------------------------------
// MI455X (gfx1250) fused ConvolutionDMax
//   k_scan : exclusive scan of sizes -> segment offsets
//   k_pack : swizzle conv weights into WMMA B-fragment order (f16)
//   k_main : per-segment  conv(w=1,2,3) + relu + ragged max-pool  via
//            v_wmma_f32_16x16x32_f16, x staged in LDS (f16, padded pitch)
//   k_fin  : tanh(P @ lin_w^T + b) via v_wmma_f32_16x16x4_f32
// ---------------------------------------------------------------------------

typedef _Float16 v16h __attribute__((ext_vector_type(16)));
typedef _Float16 h8   __attribute__((ext_vector_type(8)));
typedef float    v8f  __attribute__((ext_vector_type(8)));
typedef float    v2f  __attribute__((ext_vector_type(2)));

union U16 { v16h v; h8 h[2]; };

#define MAXROWS 66          // 64 rows max + 2 tap overhang
#define XPITCH  136         // halves per row: 128 + 8 pad -> 272B row stride (bank-conflict free)

// ------------------------------ scan ---------------------------------------
__global__ __launch_bounds__(256) void k_scan(const int* __restrict__ sizes,
                                              int* __restrict__ offs, int B) {
  __shared__ int part[256];
  const int tid = threadIdx.x;
  const int chunk = (B + 255) / 256;
  int s = 0;
  for (int j = 0; j < chunk; ++j) {
    int idx = tid * chunk + j;
    if (idx < B) s += sizes[idx];
  }
  part[tid] = s;
  __syncthreads();
  if (tid == 0) {
    int run = 0;
    for (int i = 0; i < 256; ++i) { int t = part[i]; part[i] = run; run += t; }
  }
  __syncthreads();
  int base = part[tid];
  for (int j = 0; j < chunk; ++j) {
    int idx = tid * chunk + j;
    if (idx < B) { offs[idx] = base; base += sizes[idx]; }
  }
}

// ------------------------------ weight pack --------------------------------
// packed layout: [pair(6)][ks(4)][ntile(8)][lane(32)][16 halves] -- each lane's
// 16 halves are its contiguous v16h B fragment (B is 32x16 KxN, lane<16: K=0..7,16..23,
// lane>=16: K=8..15,24..31, column N = lane&15).
__global__ __launch_bounds__(256) void k_pack(const float* __restrict__ w0,
                                              const float* __restrict__ w1,
                                              const float* __restrict__ w2,
                                              _Float16* __restrict__ packed) {
  int idx = blockIdx.x * 256 + threadIdx.x;
  if (idx >= 6 * 4 * 8 * 32 * 16) return;
  const int j    = idx & 15;
  const int lane = (idx >> 4) & 31;
  const int n    = (idx >> 9) & 7;
  const int ks   = (idx >> 12) & 3;
  const int p    = idx >> 14;                 // 0..5 : (conv,tap) pairs
  const int conv = (p == 0) ? 0 : (p < 3 ? 1 : 2);
  const int kk   = (p == 0) ? 0 : (p < 3 ? p - 1 : p - 3);
  const int taps = conv + 1;
  const int m    = n * 16 + (lane & 15);      // output channel (N)
  const int K    = ks * 32 + (j >> 3) * 16 + (lane >> 4) * 8 + (j & 7);  // input channel (K)
  const float* w = (conv == 0) ? w0 : (conv == 1) ? w1 : w2;
  packed[idx] = (_Float16)w[(m * 128 + K) * taps + kk];
}

// ------------------------------ main fused kernel --------------------------
__global__ __launch_bounds__(128) void k_main(const float* __restrict__ x,
                                              const int* __restrict__ sizes,
                                              const int* __restrict__ offs,
                                              const _Float16* __restrict__ packed,
                                              const float* __restrict__ b0,
                                              const float* __restrict__ b1,
                                              const float* __restrict__ b2,
                                              float* __restrict__ poolOut) {
  __shared__ _Float16 xl[MAXROWS * XPITCH];
  __shared__ int pool[3 * 128];

  const int seg = blockIdx.x;
  int size = sizes[seg];
  if (size > 64) size = 64;                   // defensive clamp (ref: 16..63)
  const long off = offs[seg];
  const int tid = threadIdx.x;

  pool[tid] = 0; pool[tid + 128] = 0; pool[tid + 256] = 0;

  // stage x rows into LDS as f16 (coalesced float2 reads, b32 LDS writes)
  for (int i = tid; i < size * 64; i += 128) {
    const int r = i >> 6, c = (i & 63) * 2;
    const float* xp = x + (off + r) * 128 + c;
    xl[r * XPITCH + c]     = (_Float16)xp[0];
    xl[r * XPITCH + c + 1] = (_Float16)xp[1];
  }
  for (int i = tid; i < (MAXROWS - size) * 64; i += 128) {
    const int r = size + (i >> 6), c = (i & 63) * 2;
    xl[r * XPITCH + c]     = (_Float16)0.f;
    xl[r * XPITCH + c + 1] = (_Float16)0.f;
  }
  __syncthreads();

  const int ntiles = (size + 15) >> 4;        // 1..4 t-tiles of 16 rows
  const int lane = tid & 31, wid = tid >> 5;
  const int hi = lane >> 4, lm = lane & 15;
  const float* biasP[3] = {b0, b1, b2};
  constexpr int convA[6] = {0, 1, 1, 2, 2, 2};
  constexpr int kkA[6]   = {0, 0, 1, 0, 1, 2};

  // wave -> (t-tile pair, n half); each wave computes 2 t-tiles x 2 n-tiles
  // for all 3 convs at once (3 shifted A frags shared across the 6 matmuls).
  for (int tt0 = (wid >> 1) * 2; tt0 < ntiles; tt0 += 4) {
    for (int np = 0; np < 2; ++np) {
      const int n0 = (wid & 1) * 4 + np * 2;
      v8f acc[3][2][2];
      const v8f vz = {0.f, 0.f, 0.f, 0.f, 0.f, 0.f, 0.f, 0.f};
#pragma unroll
      for (int cv = 0; cv < 3; ++cv)
#pragma unroll
        for (int mt = 0; mt < 2; ++mt)
#pragma unroll
          for (int nt = 0; nt < 2; ++nt) acc[cv][mt][nt] = vz;

#pragma unroll
      for (int ks = 0; ks < 4; ++ks) {
        // A fragments: rows (tt0+mt)*16 + shift, K-chunks ks*32+{hi*8, 16+hi*8}
        v16h A[2][3];
#pragma unroll
        for (int mt = 0; mt < 2; ++mt)
#pragma unroll
          for (int sh = 0; sh < 3; ++sh) {
            const int row = (tt0 + mt) * 16 + sh + lm;
            const _Float16* ap = &xl[row * XPITCH + ks * 32 + hi * 8];
            U16 u;
            u.h[0] = *(const h8*)(ap);        // K = ks*32 + hi*8 .. +7
            u.h[1] = *(const h8*)(ap + 16);   // K = ks*32 + 16 + hi*8 .. +7
            A[mt][sh] = u.v;
          }
#pragma unroll
        for (int p = 0; p < 6; ++p) {
#pragma unroll
          for (int nt = 0; nt < 2; ++nt) {
            const _Float16* bp =
                packed + ((((p * 4 + ks) * 8 + (n0 + nt)) * 32 + lane) << 4);
            U16 ub;
            ub.h[0] = *(const h8*)(bp);
            ub.h[1] = *(const h8*)(bp + 8);
            const v16h Bf = ub.v;
#pragma unroll
            for (int mt = 0; mt < 2; ++mt) {
              acc[convA[p]][mt][nt] = __builtin_amdgcn_wmma_f32_16x16x32_f16(
                  false, A[mt][kkA[p]], false, Bf, (short)0,
                  acc[convA[p]][mt][nt], false, false);
            }
          }
        }
      }

      // epilogue: bias + relu + ragged-validity mask + max-pool
#pragma unroll
      for (int cv = 0; cv < 3; ++cv) {
        const int lim = size - cv;            // valid t in [0, size - (w-1))
#pragma unroll
        for (int nt = 0; nt < 2; ++nt) {
          const int m = (n0 + nt) * 16 + lm;
          const float bb = biasP[cv][m];
#pragma unroll
          for (int mt = 0; mt < 2; ++mt) {
            const int tb = (tt0 + mt) * 16 + hi * 8;
            float mx = 0.f;
#pragma unroll
            for (int v = 0; v < 8; ++v) {
              float val = acc[cv][mt][nt][v] + bb;
              val = fmaxf(val, 0.f);
              if (tb + v >= lim) val = 0.f;
              mx = fmaxf(mx, val);
            }
            mx = fmaxf(mx, __shfl_xor(mx, 16, 32));
            if (hi == 0)                      // relu>=0 -> int-bit max == float max
              atomicMax(&pool[cv * 128 + m], __float_as_int(mx));
          }
        }
      }
    }
  }
  __syncthreads();

  float* po = poolOut + (long)seg * 384;
  po[tid]       = __int_as_float(pool[tid]);
  po[tid + 128] = __int_as_float(pool[tid + 128]);
  po[tid + 256] = __int_as_float(pool[tid + 256]);
}

// ------------------------------ final GEMM + tanh --------------------------
// out[8192,128] = tanh(P[8192,384] @ lin_w[128,384]^T + lin_b), fp32 WMMA.
__global__ __launch_bounds__(128) void k_fin(const float* __restrict__ P,
                                             const float* __restrict__ lw,
                                             const float* __restrict__ lb,
                                             float* __restrict__ out) {
  const int lane = threadIdx.x & 31, wid = threadIdx.x >> 5;
  const int hi = lane >> 4, lm = lane & 15;
  const long row0 = (long)blockIdx.x * 64 + wid * 16;

  v8f acc[8];
  const v8f vz = {0.f, 0.f, 0.f, 0.f, 0.f, 0.f, 0.f, 0.f};
#pragma unroll
  for (int n = 0; n < 8; ++n) acc[n] = vz;

  for (int c0 = 0; c0 < 384; c0 += 4) {
    // A 16x4: lane holds (M=lm, K=c0+2*hi, c0+2*hi+1) -> contiguous float2
    const v2f Af = *(const v2f*)(P + (row0 + lm) * 384 + c0 + hi * 2);
#pragma unroll
    for (int n = 0; n < 8; ++n) {
      // B 4x16: lane holds (K=c0+2*hi..+1, N=lm); B[k][n] = lin_w[n][k]
      const v2f Bf = *(const v2f*)(lw + (n * 16 + lm) * 384 + c0 + hi * 2);
      acc[n] = __builtin_amdgcn_wmma_f32_16x16x4_f32(
          false, Af, false, Bf, (short)0, acc[n], false, false);
    }
  }
#pragma unroll
  for (int n = 0; n < 8; ++n) {
    const int col = n * 16 + lm;
    const float bb = lb[col];
#pragma unroll
    for (int v = 0; v < 8; ++v) {
      const long r = row0 + v + hi * 8;
      out[r * 128 + col] = tanhf(acc[n][v] + bb);
    }
  }
}

// ------------------------------ launcher -----------------------------------
extern "C" void kernel_launch(void* const* d_in, const int* in_sizes, int n_in,
                              void* d_out, int out_size, void* d_ws, size_t ws_size,
                              hipStream_t stream) {
  const float* x     = (const float*)d_in[0];
  const int*   sizes = (const int*)d_in[1];
  const float* w0 = (const float*)d_in[2];
  const float* b0 = (const float*)d_in[3];
  const float* w1 = (const float*)d_in[4];
  const float* b1 = (const float*)d_in[5];
  const float* w2 = (const float*)d_in[6];
  const float* b2 = (const float*)d_in[7];
  const float* lw = (const float*)d_in[8];
  const float* lb = (const float*)d_in[9];
  const int B = in_sizes[1];                  // 8192 segments

  char* ws = (char*)d_ws;
  int*       offs    = (int*)ws;                               // 32 KB
  _Float16*  packed  = (_Float16*)(ws + 32768);                // 192 KB
  float*     poolOut = (float*)(ws + 32768 + 196608);          // B*384*4

  k_scan<<<1, 256, 0, stream>>>(sizes, offs, B);
  k_pack<<<(6 * 4 * 8 * 32 * 16 + 255) / 256, 256, 0, stream>>>(w0, w1, w2, packed);
  k_main<<<B, 128, 0, stream>>>(x, sizes, offs, packed, b0, b1, b2, poolOut);
  k_fin<<<B / 64, 128, 0, stream>>>(poolOut, lw, lb, (float*)d_out);
}
